// AttentionPolicyHead_87514253623540
// MI455X (gfx1250) — matrix operand order repaired and checked by reading, not compiled, not run on hardware
//
#include <hip/hip_runtime.h>
#include <math.h>

#define SQN    81
#define DMODEL 256
#define PPOL   3849
#define NDROPQ 7
#define MPAD   96        // 81 padded to 6 tiles of 16
#define MPADA  112       // 96 + 16 extra rows holding drop_queries (7 used)
#define KTILE  32
#define NTHREADS 256
#define NWAVES 8
#define SCALEF 0.0625f   // 1/sqrt(256)
#define WELEMS (DMODEL*DMODEL)   // 65536 elements per weight matrix

typedef __attribute__((ext_vector_type(16))) __bf16 v16bf;
typedef __attribute__((ext_vector_type(8)))  float  v8f;

static __device__ __forceinline__ v8f wmma_bf16(v16bf a, v16bf b, v8f c) {
  // D(f32 16x16) = A(bf16 16x32) * B(bf16 32x16) + C
  return __builtin_amdgcn_wmma_f32_16x16x32_bf16(false, a, false, b, (short)0, c,
                                                 false, false);
}

// A fragment 16x32 from row-major LDS tile (pitch DMODEL).
// lane L: M = L%16, h = L/16; halves 0-15 hold K = 16h+0..15 (contiguous).
static __device__ __forceinline__ v16bf load_A(const __bf16* base, int lane) {
  const int m = lane & 15, h = lane >> 4;
  const __bf16* row = base + m * DMODEL;
  v16bf a;
#pragma unroll
  for (int j = 0; j < 16; ++j) {
    a[j] = row[16*h + j];
  }
  return a;
}

// B fragment 32x16 loaded directly from the pre-swizzled bf16 weight image in
// global memory (L2-resident). Tile (kc,nt) occupies 512 halves; lane L's 16
// halves are contiguous at lane*16 -> two coalesced global_load_b128 per lane.
static __device__ __forceinline__ v16bf load_Bg(const __bf16* Wf, int kc, int nt, int lane) {
  const __bf16* src = Wf + ((kc * 16 + nt) << 9) + (lane << 4);
  return *(const v16bf*)src;
}

// B fragment where logical B = X^T, X row-major in LDS: B(k,n) = X[n0+n][k0+k].
// Each lane reads 16 contiguous bf16 from one row of X.
static __device__ __forceinline__ v16bf load_Bt(const __bf16* base, int lane) {
  const int n = lane & 15, h = lane >> 4;
  const __bf16* src = base + n * DMODEL + 16 * h;
  v16bf b;
#pragma unroll
  for (int i = 0; i < 16; ++i) b[i] = src[i];
  return b;
}

// C = A(96x256, LDS bf16) @ W(pre-swizzled bf16, global/L2) + bias,
// optional mish, result written bf16 to ldsOut (96x256). Barrier-free.
static __device__ __forceinline__ void gemm256(
    const __bf16* __restrict__ A,
    const __bf16* __restrict__ Wf,     // fragment-swizzled bf16 weight image
    const float*  __restrict__ bias,
    __bf16* __restrict__ ldsOut,
    bool act, int tid)
{
  const int lane = tid & 31;
  const int wv   = tid >> 5;
  const int nt0  = 2 * wv;           // this wave owns N-tiles nt0, nt0+1

  const v8f vzero = {0.f,0.f,0.f,0.f,0.f,0.f,0.f,0.f};
  v8f acc[6][2];
#pragma unroll
  for (int m = 0; m < 6; ++m) { acc[m][0] = vzero; acc[m][1] = vzero; }

#pragma unroll
  for (int kc = 0; kc < 8; ++kc) {
    v16bf b0 = load_Bg(Wf, kc, nt0,     lane);
    v16bf b1 = load_Bg(Wf, kc, nt0 + 1, lane);
    if (kc < 7) {
      __builtin_prefetch(Wf + (((kc+1) * 16 + nt0) << 9) + (lane << 4), 0, 0);
    }
#pragma unroll
    for (int m = 0; m < 6; ++m) {
      v16bf a = load_A(A + m*16*DMODEL + kc*KTILE, lane);
      acc[m][0] = wmma_bf16(a, b0, acc[m][0]);
      acc[m][1] = wmma_bf16(a, b1, acc[m][1]);
    }
  }

  // epilogue: bias (+ mish) -> bf16 LDS.  C/D layout: lane N = L%16, rows r+8h.
  const int n = lane & 15, h = lane >> 4;
#pragma unroll
  for (int m = 0; m < 6; ++m)
#pragma unroll
    for (int nn = 0; nn < 2; ++nn) {
      const int col = (nt0 + nn) * 16 + n;
      const float bv = bias[col];
#pragma unroll
      for (int r = 0; r < 8; ++r) {
        const int row = m*16 + r + 8*h;
        float v = acc[m][nn][r] + bv;
        if (act) {
          float sp = (v > 20.f) ? v : log1pf(__expf(v));   // softplus
          v = v * tanhf(sp);                               // mish
        }
        ldsOut[row*DMODEL + col] = (__bf16)v;
      }
    }
}

// One-time weight prep: f32 (256x256 row-major, in x out) -> bf16 image in the
// exact per-lane B-fragment order: tile t = kc*16+nt (512 halves), within-tile
// j = L*16 + i with n = nt*16 + L%16, k = kc*32 + 16*(L/16) + i.
__global__ void __launch_bounds__(NTHREADS)
prep_weights(const float* __restrict__ We, const float* __restrict__ Wq,
             const float* __restrict__ Wk, __bf16* __restrict__ wsw)
{
  const int id = blockIdx.x * NTHREADS + threadIdx.x;   // 0 .. 3*65536-1
  if (id >= 3 * WELEMS) return;
  const float* W = (id < WELEMS) ? We : (id < 2*WELEMS ? Wq : Wk);
  const int rem  = id & (WELEMS - 1);
  const int tile = rem >> 9;          // / 512
  const int j    = rem & 511;
  const int L    = j >> 4, i = j & 15;
  const int kc   = tile >> 4, nt = tile & 15;
  const int n    = nt * 16 + (L & 15);
  const int k    = kc * 32 + 16 * (L >> 4) + i;
  wsw[id] = (__bf16)W[k * DMODEL + n];
}

__global__ void __launch_bounds__(NTHREADS)
apk_fused(const float* __restrict__ x,
          const __bf16* __restrict__ wsw,     // [We|Wq|Wk] swizzled bf16
          const float* __restrict__ be, const float* __restrict__ bq,
          const float* __restrict__ bk,
          const float* __restrict__ Wp, const float* __restrict__ bp,
          const float* __restrict__ dq,
          const int* __restrict__ rawIdx, const int* __restrict__ polIdx,
          float* __restrict__ out)
{
  extern __shared__ char smem[];
  __bf16* bufA = (__bf16*)smem;                 // rows 0-95: x -> Q, rows 96-111: dq
  __bf16* bufB = bufA + MPADA*DMODEL;           // pe -> board (f32)
  __bf16* bufC = bufB + MPAD*DMODEL;            // K
  float*  aux  = (float*)(bufC + MPAD*DMODEL);  // [0..95] prow, [96..96+567) drop
  float*  board = (float*)bufB;                 // 96x96 f32 (36.9 KB < 49.2 KB)

  const int b    = blockIdx.x;
  const int tid  = threadIdx.x;
  const int lane = tid & 31;
  const int wv   = tid >> 5;

  // stage x_b (f32 -> bf16), zero pad rows 81..95
  const float4* xb = (const float4*)(x + (size_t)b * SQN * DMODEL);
  for (int i = tid; i < (SQN*DMODEL)/4; i += NTHREADS) {
    float4 v = xb[i];
    const int o = 4 * i;
    bufA[o+0] = (__bf16)v.x; bufA[o+1] = (__bf16)v.y;
    bufA[o+2] = (__bf16)v.z; bufA[o+3] = (__bf16)v.w;
  }
  for (int i = SQN*DMODEL + tid; i < MPAD*DMODEL; i += NTHREADS)
    bufA[i] = (__bf16)0.f;
  // stage drop_queries into rows 96..102 (f32 -> bf16), zero rows 103..111
  {
    const float4* dq4 = (const float4*)dq;
    __bf16* dst = bufA + MPAD*DMODEL;
    for (int i = tid; i < (NDROPQ*DMODEL)/4; i += NTHREADS) {
      float4 v = dq4[i];
      const int o = 4 * i;
      dst[o+0] = (__bf16)v.x; dst[o+1] = (__bf16)v.y;
      dst[o+2] = (__bf16)v.z; dst[o+3] = (__bf16)v.w;
    }
    for (int i = NDROPQ*DMODEL + tid; i < 16*DMODEL; i += NTHREADS)
      dst[i] = (__bf16)0.f;
  }
  __syncthreads();

  gemm256(bufA, wsw,              be, bufB, true,  tid);  // pe = mish(x@We+be)
  __syncthreads();
  gemm256(bufB, wsw +   WELEMS,   bq, bufA, false, tid);  // Q = pe@Wq+bq (rows 0-95)
  __syncthreads();
  gemm256(bufB, wsw + 2*WELEMS,   bk, bufC, false, tid);  // K = pe@Wk+bk
  __syncthreads();

  // promo row: prow[k] = K[k] . Wp + bp  (small, stays on VALU)
  if (tid < SQN) {
    float s = 0.f;
#pragma unroll 4
    for (int d = 0; d < DMODEL; ++d) s += (float)bufC[tid*DMODEL + d] * Wp[d];
    aux[tid] = s + bp[0];
  }

  // [board | drop] = scale * [Q ; dq] @ K^T : 7x6 = 42 tiles across 8 waves.
  // M-tiles 0..5 -> board (96x96 LDS), M-tile 6 -> drop logits (7x81 in aux).
  for (int t = wv; t < 42; t += NWAVES) {
    const int mt = t / 6, nt = t % 6;
    v8f acc = {0.f,0.f,0.f,0.f,0.f,0.f,0.f,0.f};
#pragma unroll
    for (int kc = 0; kc < 8; ++kc) {
      v16bf a  = load_A (bufA + mt*16*DMODEL + kc*KTILE, lane);
      v16bf bt = load_Bt(bufC + nt*16*DMODEL + kc*KTILE, lane);
      acc = wmma_bf16(a, bt, acc);
    }
    const int n = lane & 15, h = lane >> 4;
    if (mt < 6) {
#pragma unroll
      for (int r = 0; r < 8; ++r)
        board[(mt*16 + r + 8*h)*MPAD + nt*16 + n] = acc[r] * SCALEF;
    } else {
      const int col = nt*16 + n;
#pragma unroll
      for (int r = 0; r < 8; ++r) {
        const int row = r + 8*h;              // dq row 0..15, valid < 7
        if (row < NDROPQ && col < SQN)
          aux[96 + row*SQN + col] = acc[r] * SCALEF;
      }
    }
  }
  __syncthreads();

  // scatter: pol_idx is a permutation of P, so every slot is written exactly once
  float* og = out + (size_t)b * PPOL;
  for (int j = tid; j < PPOL; j += NTHREADS) {
    const int r = rawIdx[j];
    float v;
    if (r < SQN*SQN) {
      v = board[(r / SQN) * MPAD + (r % SQN)];
    } else if (r < 2*SQN*SQN) {
      const int rr = r - SQN*SQN;
      v = board[(rr / SQN) * MPAD + (rr % SQN)] + aux[rr % SQN];
    } else {
      v = aux[96 + (r - 2*SQN*SQN)];
    }
    og[polIdx[j]] = v;
  }
}

extern "C" void kernel_launch(void* const* d_in, const int* in_sizes, int n_in,
                              void* d_out, int out_size, void* d_ws, size_t ws_size,
                              hipStream_t stream) {
  (void)n_in; (void)ws_size; (void)out_size;
  const float* x   = (const float*)d_in[0];
  const float* We  = (const float*)d_in[1];
  const float* be  = (const float*)d_in[2];
  const float* Wq  = (const float*)d_in[3];
  const float* bq  = (const float*)d_in[4];
  const float* Wk  = (const float*)d_in[5];
  const float* bk  = (const float*)d_in[6];
  const float* Wp  = (const float*)d_in[7];
  const float* bp  = (const float*)d_in[8];
  const float* dq  = (const float*)d_in[9];
  const int* rawIdx = (const int*)d_in[10];
  const int* polIdx = (const int*)d_in[11];
  float* out = (float*)d_out;
  __bf16* wsw = (__bf16*)d_ws;                 // 3 * 65536 bf16 = 384 KB

  const int Bn = in_sizes[0] / (SQN * DMODEL);

  // one-time (per launch) weight convert + swizzle into d_ws
  prep_weights<<<(3 * WELEMS + NTHREADS - 1) / NTHREADS, NTHREADS, 0, stream>>>(
      We, Wq, Wk, wsw);

  const size_t smem = (size_t)((MPADA + 2*MPAD) * DMODEL) * 2    // bf16 bufs
                    + (size_t)(96 + NDROPQ * SQN + 1) * 4;       // aux f32
  hipFuncSetAttribute(reinterpret_cast<const void*>(apk_fused),
                      hipFuncAttributeMaxDynamicSharedMemorySize, (int)smem);
  apk_fused<<<Bn, NTHREADS, smem, stream>>>(x, wsw, be, bq, bk, Wp, bp, dq,
                                            rawIdx, polIdx, out);
}
